// BWENet_81466939670627
// MI455X (gfx1250) — matrix operand order repaired and checked by reading, not compiled, not run on hardware
//
#include <hip/hip_runtime.h>
#include <hip/hip_bf16.h>

typedef _Float16 v16h __attribute__((ext_vector_type(16)));
typedef _Float16 v8h  __attribute__((ext_vector_type(8)));
typedef float    v8f  __attribute__((ext_vector_type(8)));

#define KS 15
#define NFR 1000
#define CD  128
#define PI_F 3.14159265358979323846f
#define GA_F 1.3815510557964274f

__device__ __forceinline__ v8f wmma_f16(v16h a, v16h b, v8f c) {
  return __builtin_amdgcn_wmma_f32_16x16x32_f16(false, a, false, b, (short)0, c, false, false);
}

__device__ __forceinline__ float fast_sigmoid(float v) {
  return __builtin_amdgcn_rcpf(1.f + __expf(-v));          // v_exp_f32 + v_rcp_f32
}
__device__ __forceinline__ float fast_tanh(float v) {
  float e2 = __expf(-2.f * v);
  return (1.f - e2) * __builtin_amdgcn_rcpf(1.f + e2);
}

// ---------------- feature net convs (scalar, small) ----------------

__global__ __launch_bounds__(256) void k_conv1(const float* __restrict__ feats,
                                               const float* __restrict__ w,
                                               const float* __restrict__ bias,
                                               float* __restrict__ out) {
  int idx = blockIdx.x * blockDim.x + threadIdx.x;
  if (idx >= 32 * 128 * 500) return;
  int t = idx % 500; int o = (idx / 500) % 128; int b = idx / (500 * 128);
  float s = bias[o];
  #pragma unroll
  for (int j = 0; j < 3; ++j) {
    int tt = t - 2 + j;
    if (tt < 0) continue;
    const float* fp = feats + ((long)b * 500 + tt) * 84;
    const float* wp = w + (long)o * 84 * 3 + j;
    for (int c = 0; c < 84; ++c) s += fp[c] * wp[c * 3];
  }
  out[idx] = tanhf(s);
}

__global__ __launch_bounds__(256) void k_conv2(const float* __restrict__ in,
                                               const float* __restrict__ w,
                                               const float* __restrict__ bias,
                                               float* __restrict__ out) {
  int idx = blockIdx.x * blockDim.x + threadIdx.x;
  if (idx >= 32 * 128 * 500) return;
  int t = idx % 500; int o = (idx / 500) % 128; int b = idx / (500 * 128);
  float s = bias[o];
  #pragma unroll
  for (int j = 0; j < 3; ++j) {
    int tt = t - 2 + j;
    if (tt < 0) continue;
    const float* ip = in + (long)b * 128 * 500 + tt;   // [ (b*128+c)*500 + tt ]
    const float* wp = w + (long)o * 128 * 3 + j;
    for (int c = 0; c < 128; ++c) s += ip[c * 500] * wp[c * 3];
  }
  out[idx] = tanhf(s);
}

// transposed conv (time x2) : cbuf[(b*1000+tt)*128+o] = tanh( sum_c c2[b,c,t]*tw[c,o,j] + tb[o] )
__global__ __launch_bounds__(256) void k_tconv(const float* __restrict__ c2,
                                               const float* __restrict__ tw,
                                               const float* __restrict__ tb,
                                               float* __restrict__ cbuf) {
  int idx = blockIdx.x * blockDim.x + threadIdx.x;
  if (idx >= 32 * 128 * 1000) return;
  int tt = idx % 1000; int o = (idx / 1000) % 128; int b = idx / (1000 * 128);
  int t = tt >> 1, j = tt & 1;
  float s = tb[o];
  const float* ip = c2 + (long)b * 128 * 500 + t;
  for (int c = 0; c < 128; ++c) s += ip[c * 500] * tw[(long)c * 256 + o * 2 + j];
  cbuf[((long)b * 1000 + tt) * 128 + o] = tanhf(s);
}

// ---------------- WMMA GEMM, 8 tile-waves per workgroup ----------------
// out[m,n] = sum_k A[m,k]*W[n,k] + bias[n];  K=128 -> 4 wmma 16x16x32 steps per tile
__global__ __launch_bounds__(256) void k_gemm_xw(const float* __restrict__ A,
                                                 const float* __restrict__ W,
                                                 const float* __restrict__ bias,
                                                 float* __restrict__ out,
                                                 int ldo, int ntn, int ntiles) {
  const int K = 128;
  int tile = blockIdx.x * 8 + (threadIdx.x >> 5);      // wave-uniform
  if (tile >= ntiles) return;                           // whole-wave exit: EXEC stays full at wmma
  int mt = tile / ntn, nt = tile % ntn;
  int lane = threadIdx.x & 31;
  int ln16 = lane & 15, half = lane >> 4;
  int m = mt * 16 + ln16;
  int n = nt * 16 + ln16;
  v8f acc;
  float bv = bias[n];
  #pragma unroll
  for (int r = 0; r < 8; ++r) acc[r] = bv;
  #pragma unroll
  for (int kt = 0; kt < 4; ++kt) {
    v16h a, bm;
    int k0 = kt * 32 + half * 16;
    const float* ap = A + (long)m * K + k0;
    #pragma unroll
    for (int i = 0; i < 8; ++i) { a[i] = (_Float16)ap[i]; a[8 + i] = (_Float16)ap[8 + i]; }
    const float* wp = W + (long)n * K + kt * 32 + half * 16;
    #pragma unroll
    for (int i = 0; i < 16; ++i) bm[i] = (_Float16)wp[i];
    acc = wmma_f16(a, bm, acc);
  }
  #pragma unroll
  for (int r = 0; r < 8; ++r)
    out[(long)(mt * 16 + r + half * 8) * ldo + nt * 16 + ln16] = acc[r];
}

// adaconv kernel generation: N=32 (cols 0..29 = wk rows, 30..30+cout-1 = wg rows)
__global__ __launch_bounds__(256) void k_gemm_kgen(const float* __restrict__ A,
                                                   const float* __restrict__ wk,
                                                   const float* __restrict__ bk,
                                                   const float* __restrict__ wg,
                                                   const float* __restrict__ bg,
                                                   int cout,
                                                   float* __restrict__ out,
                                                   int ntiles) {
  const int K = 128;
  int tile = blockIdx.x * 8 + (threadIdx.x >> 5);
  if (tile >= ntiles) return;
  int mt = tile >> 1, nt = tile & 1;                    // ntn == 2
  int lane = threadIdx.x & 31;
  int ln16 = lane & 15, half = lane >> 4;
  int m = mt * 16 + ln16;
  int n = nt * 16 + ln16;
  const float* wrow = wk;       // safe dummy
  float bv = 0.f, msk = 0.f;
  if (n < 30)            { wrow = wk + (long)n * K;        bv = bk[n];      msk = 1.f; }
  else if (n < 30 + cout){ wrow = wg + (long)(n - 30) * K; bv = bg[n - 30]; msk = 1.f; }
  v8f acc;
  #pragma unroll
  for (int r = 0; r < 8; ++r) acc[r] = bv;
  #pragma unroll
  for (int kt = 0; kt < 4; ++kt) {
    v16h a, bm;
    int k0 = kt * 32 + half * 16;
    const float* ap = A + (long)m * K + k0;
    #pragma unroll
    for (int i = 0; i < 8; ++i) { a[i] = (_Float16)ap[i]; a[8 + i] = (_Float16)ap[8 + i]; }
    const float* wp = wrow + kt * 32 + half * 16;
    #pragma unroll
    for (int i = 0; i < 16; ++i) bm[i] = (_Float16)(msk * wp[i]);
    acc = wmma_f16(a, bm, acc);
  }
  #pragma unroll
  for (int r = 0; r < 8; ++r)
    out[(long)(mt * 16 + r + half * 8) * 32 + nt * 16 + ln16] = acc[r];
}

__global__ __launch_bounds__(256) void k_f32tof16(const float* __restrict__ in,
                                                  _Float16* __restrict__ out, int n) {
  int i = blockIdx.x * blockDim.x + threadIdx.x;
  if (i < n) out[i] = (_Float16)in[i];
}

// ---------------- GRU: single workgroup, 12 waves, whh B-tiles register-resident ----------------
__global__ __launch_bounds__(384, 1) void k_gru(const float* __restrict__ xw,
                                                const _Float16* __restrict__ whh16,
                                                const float* __restrict__ bhh,
                                                float* __restrict__ cf) {
  __shared__ __align__(16) _Float16 h16[32 * 128];
  __shared__ __align__(16) float    gh[32 * 384];
  int tid = threadIdx.x;
  int w = tid >> 5, lane = tid & 31;
  int ln16 = lane & 15, half = lane >> 4;
  int n0 = w * 32;

  // preload B tiles (whh rows n0..n0+31, K=128) into registers, WMMA B layout
  v16h Breg[2][4];
  #pragma unroll
  for (int ntl = 0; ntl < 2; ++ntl) {
    int n = n0 + ntl * 16 + ln16;
    #pragma unroll
    for (int kt = 0; kt < 4; ++kt) {
      const _Float16* wp = whh16 + (long)n * 128 + kt * 32 + half * 16;
      v16h bv;
      #pragma unroll
      for (int i = 0; i < 16; ++i) bv[i] = wp[i];
      Breg[ntl][kt] = bv;
    }
  }
  float bh0 = bhh[n0 + ln16], bh1 = bhh[n0 + 16 + ln16];

  for (int i = tid; i < 32 * 128; i += 384) h16[i] = (_Float16)0.f;
  __syncthreads();

  for (int t = 0; t < NFR; ++t) {
    v8f acc[2][2];
    #pragma unroll
    for (int r = 0; r < 8; ++r) {
      acc[0][0][r] = bh0; acc[1][0][r] = bh0;
      acc[0][1][r] = bh1; acc[1][1][r] = bh1;
    }
    #pragma unroll
    for (int mt2 = 0; mt2 < 2; ++mt2) {
      int m = mt2 * 16 + ln16;
      #pragma unroll
      for (int kt = 0; kt < 4; ++kt) {
        int k0 = kt * 32 + half * 16;
        v8h lo = *(const v8h*)&h16[m * 128 + k0];
        v8h hi = *(const v8h*)&h16[m * 128 + k0 + 8];
        v16h a;
        #pragma unroll
        for (int i = 0; i < 8; ++i) { a[i] = lo[i]; a[8 + i] = hi[i]; }
        acc[mt2][0] = wmma_f16(a, Breg[0][kt], acc[mt2][0]);
        acc[mt2][1] = wmma_f16(a, Breg[1][kt], acc[mt2][1]);
      }
    }
    #pragma unroll
    for (int mt2 = 0; mt2 < 2; ++mt2)
      #pragma unroll
      for (int ntl = 0; ntl < 2; ++ntl)
        #pragma unroll
        for (int r = 0; r < 8; ++r)
          gh[(mt2 * 16 + r + half * 8) * 384 + n0 + ntl * 16 + ln16] = acc[mt2][ntl][r];
    __syncthreads();

    for (int idx = tid; idx < 32 * 128; idx += 384) {
      int b = idx >> 7, j = idx & 127;
      const float* xb = xw + ((long)b * NFR + t) * 384;
      if (t + 1 < NFR) __builtin_prefetch(xb + 384 + j, 0, 1);   // global_prefetch_b8: next step's row
      float xr = xb[j], xz = xb[128 + j], xn = xb[256 + j];
      float hr = gh[b * 384 + j], hz = gh[b * 384 + 128 + j], hn = gh[b * 384 + 256 + j];
      float hp = (float)h16[idx];
      float r = fast_sigmoid(xr + hr);
      float z = fast_sigmoid(xz + hz);
      float nn = fast_tanh(xn + r * hn);
      float hnew = (1.f - z) * nn + z * hp;
      cf[((long)b * NFR + t) * 128 + j] = hnew;
      h16[idx] = (_Float16)hnew;
    }
    __syncthreads();
  }
}

// ---------------- polyphase resamplers ----------------

__global__ __launch_bounds__(256) void k_up2(const float* __restrict__ x,
                                             const float* __restrict__ h,
                                             float* __restrict__ y) {
  int idx = blockIdx.x * blockDim.x + threadIdx.x;
  if (idx >= 32 * 160000) return;
  int n = idx % 160000, b = idx / 160000;
  const float* xb = x + (long)b * 80000;
  float s = 0.f;
  for (int k = 0; k < 31; ++k) {
    int j = n + k - 15;
    if (j < 0 || (j & 1)) continue;
    int xi = j >> 1;
    if (xi >= 80000) break;
    s += h[k] * xb[xi];
  }
  y[idx] = s;
}

// up by 3 (47-tap) then take every 2nd sample (96k path fused away)
__global__ __launch_bounds__(256) void k_up3half(const float* __restrict__ x,
                                                 const float* __restrict__ h,
                                                 float* __restrict__ y) {
  int idx = blockIdx.x * blockDim.x + threadIdx.x;
  if (idx >= 32 * 240000) return;
  int n = idx % 240000, b = idx / 240000;
  const float* xb = x + (long)b * 160000;
  int m = 2 * n;
  float s = 0.f;
  for (int k = 0; k < 47; ++k) {
    int j = m + k - 23;
    if (j < 0 || (j % 3)) continue;
    int xi = j / 3;
    if (xi >= 160000) break;
    s += h[k] * xb[xi];
  }
  y[idx] = s;
}

// ---------------- adaconv: normalize/gate kernels, then fused apply + overlap-add ----------------

__global__ __launch_bounds__(256) void k_kpost(const float* __restrict__ raw,
                                               int cout, int cin,
                                               float* __restrict__ kf) {
  int m = blockIdx.x * blockDim.x + threadIdx.x;
  if (m >= 32 * NFR) return;
  const float* r = raw + (long)m * 32;
  int nk = cout * cin * KS;
  float* o = kf + (long)m * nk;
  for (int co = 0; co < cout; ++co) {
    float g = __expf(GA_F * fast_tanh(r[30 + co]));
    for (int ci = 0; ci < cin; ++ci) {
      const float* kv = r + (co * cin + ci) * KS;
      float ss = 0.f;
      #pragma unroll
      for (int kk = 0; kk < KS; ++kk) ss += kv[kk] * kv[kk];
      float sc = g / (1e-6f + sqrtf(ss));
      #pragma unroll
      for (int kk = 0; kk < KS; ++kk) o[(co * cin + ci) * KS + kk] = kv[kk] * sc;
    }
  }
}

// one block per (b,frame); computes this frame's outputs AND the previous frame's tail
// (same input samples, previous frame's kernels) so overlap-add needs no frame buffer.
__global__ __launch_bounds__(256) void k_adaconv(const float* __restrict__ x0,
                                                 const float* __restrict__ x1,
                                                 const float* __restrict__ kf,
                                                 int cin, int cout, int fs, int ov,
                                                 float* __restrict__ o0,
                                                 float* __restrict__ o1) {
  __shared__ float xs[2 * (240 + KS - 1)];
  __shared__ float kc[30], kp[30];
  int bf = blockIdx.x;
  int b = bf / NFR, f = bf % NFR;
  int W = fs + KS - 1;
  long sigN = (long)NFR * fs;
  for (int ci = 0; ci < cin; ++ci) {
    const float* xin = ci ? x1 : x0;
    for (int i = threadIdx.x; i < W; i += blockDim.x) {
      long g = (long)f * fs + i - (KS - 1);
      xs[ci * W + i] = (g >= 0 && g < sigN) ? xin[(long)b * sigN + g] : 0.f;
    }
  }
  int nk = cout * cin * KS;
  if ((int)threadIdx.x < nk) {
    long m = (long)b * NFR + f;
    kc[threadIdx.x] = kf[m * nk + threadIdx.x];
    kp[threadIdx.x] = (f > 0) ? kf[(m - 1) * nk + threadIdx.x] : 0.f;
  }
  __syncthreads();
  for (int s = threadIdx.x; s < fs; s += blockDim.x) {
    float w1 = 0.f, w2 = 0.f;
    if (s < ov) {
      float c = __cosf((s + 0.5f) * PI_F / (float)ov);
      w1 = 0.5f - 0.5f * c;    // win1[s]
      w2 = 0.5f + 0.5f * c;    // win2[s]
    }
    for (int co = 0; co < cout; ++co) {
      float acc = 0.f, accp = 0.f;
      for (int ci = 0; ci < cin; ++ci) {
        const float* xp  = &xs[ci * W + s];
        const float* kcp = &kc[(co * cin + ci) * KS];
        const float* kpp = &kp[(co * cin + ci) * KS];
        #pragma unroll
        for (int kk = 0; kk < KS; ++kk) {
          acc  += xp[kk] * kcp[kk];
          accp += xp[kk] * kpp[kk];
        }
      }
      float val = (s < ov) ? (w1 * acc + w2 * accp) : acc;
      float* op = co ? o1 : o0;
      op[(long)b * sigN + (long)f * fs + s] = val;
    }
  }
}

// ---------------- td_shaper, fused per-frame (recomputes 2 neighbor env/conv1 vectors) ----------------
__global__ __launch_bounds__(256) void k_tdshaper(const float* __restrict__ xin,
                                                  const float* __restrict__ cf,
                                                  const float* __restrict__ w1,
                                                  const float* __restrict__ b1,
                                                  const float* __restrict__ w2,
                                                  const float* __restrict__ b2,
                                                  int fs, int pool,
                                                  float* __restrict__ out) {
  __shared__ float fv[3][152];
  __shared__ float a1[2][240];
  __shared__ float esc[32];
  int ne = fs / pool;          // 20
  int CH = CD + ne + 1;        // 149
  int bf = blockIdx.x;
  int b = bf / NFR, f = bf % NFR;
  long sigN = (long)NFR * fs;
  const float* xb = xin + (long)b * sigN;

  for (int slot = 0; slot < 3; ++slot) {
    int g = f - 2 + slot;                       // uniform across block
    if (g < 0) {
      for (int i = threadIdx.x; i < CH; i += blockDim.x) fv[slot][i] = 0.f;
    } else {
      for (int i = threadIdx.x; i < CD; i += blockDim.x)
        fv[slot][i] = cf[((long)b * NFR + g) * CD + i];
      if ((int)threadIdx.x < ne) {
        float m = 0.f;
        for (int p = 0; p < pool; ++p) m += fabsf(xb[(long)g * fs + threadIdx.x * pool + p]);
        m /= (float)pool;
        esc[threadIdx.x] = logf(m + 1.52587890625e-05f);   // 2^-16
      }
      __syncthreads();
      if (threadIdx.x == 0) {
        float s = 0.f;
        for (int i = 0; i < ne; ++i) s += esc[i];
        esc[ne] = s / (float)ne;
      }
      __syncthreads();
      if ((int)threadIdx.x < ne) fv[slot][CD + threadIdx.x] = esc[threadIdx.x] - esc[ne];
      if (threadIdx.x == 0)      fv[slot][CD + ne] = esc[ne];
    }
    __syncthreads();
  }

  // conv1 (k=2, causal) at t=f-1 (slots 0,1) and t=f (slots 1,2), leaky 0.2
  for (int o = threadIdx.x; o < fs; o += blockDim.x) {
    const float* wr = w1 + (long)o * CH * 2;
    float s0 = b1[o], s1 = b1[o];
    for (int c = 0; c < CH; ++c) {
      float wA = wr[c * 2], wB = wr[c * 2 + 1];
      s0 += fv[0][c] * wA + fv[1][c] * wB;
      s1 += fv[1][c] * wA + fv[2][c] * wB;
    }
    a1[0][o] = s0 > 0.f ? s0 : 0.2f * s0;
    a1[1][o] = s1 > 0.f ? s1 : 0.2f * s1;
  }
  __syncthreads();
  // conv2 (k=2, causal) at t=f, exp gain, multiply signal
  for (int o = threadIdx.x; o < fs; o += blockDim.x) {
    const float* wr = w2 + (long)o * fs * 2;
    float s = b2[o];
    for (int c = 0; c < fs; ++c) s += a1[0][c] * wr[c * 2] + a1[1][c] * wr[c * 2 + 1];
    float a = __expf(s);
    out[(long)b * sigN + (long)f * fs + o] = a * xb[(long)f * fs + o];
  }
}

// ---------------- host-side orchestration ----------------

extern "C" void kernel_launch(void* const* d_in, const int* in_sizes, int n_in,
                              void* d_out, int out_size, void* d_ws, size_t ws_size,
                              hipStream_t stream) {
  (void)in_sizes; (void)n_in; (void)out_size; (void)ws_size;
  const float* x        = (const float*)d_in[0];
  const float* features = (const float*)d_in[1];
  const float* c1w = (const float*)d_in[2];  const float* c1b = (const float*)d_in[3];
  const float* c2w = (const float*)d_in[4];  const float* c2b = (const float*)d_in[5];
  const float* tw  = (const float*)d_in[6];  const float* tb  = (const float*)d_in[7];
  const float* gwih = (const float*)d_in[8]; const float* gwhh = (const float*)d_in[9];
  const float* gbih = (const float*)d_in[10]; const float* gbhh = (const float*)d_in[11];
  const float* af_wk[4] = { (const float*)d_in[12], (const float*)d_in[16], (const float*)d_in[20], (const float*)d_in[24] };
  const float* af_bk[4] = { (const float*)d_in[13], (const float*)d_in[17], (const float*)d_in[21], (const float*)d_in[25] };
  const float* af_wg[4] = { (const float*)d_in[14], (const float*)d_in[18], (const float*)d_in[22], (const float*)d_in[26] };
  const float* af_bg[4] = { (const float*)d_in[15], (const float*)d_in[19], (const float*)d_in[23], (const float*)d_in[27] };
  const float* ts1_w1 = (const float*)d_in[28]; const float* ts1_b1 = (const float*)d_in[29];
  const float* ts1_w2 = (const float*)d_in[30]; const float* ts1_b2 = (const float*)d_in[31];
  const float* ts2_w1 = (const float*)d_in[32]; const float* ts2_b1 = (const float*)d_in[33];
  const float* ts2_w2 = (const float*)d_in[34]; const float* ts2_b2 = (const float*)d_in[35];
  const float* up2_h  = (const float*)d_in[36]; const float* up32_h = (const float*)d_in[37];
  float* dout = (float*)d_out;

  // workspace arena (floats), hand-aliased by lifetime
  float* ws = (float*)d_ws;
  float* cf       = ws;                      // 4,096,000  (persistent)
  float* xw       = ws + 4096000;            // R1: 15,360,000
  float* y32a0    = ws + 4096000;            //   reuse after GRU
  float* y32a1    = ws + 4096000 + 5120000;
  float* y48a0    = ws + 4096000;
  float* y48a1    = ws + 4096000 + 7680000;
  float* c1out    = ws + 19456000;           // R2: 7,680,000
  float* c2out    = ws + 19456000 + 2048000;
  float* y32up    = ws + 19456000;           //   reuse
  float* y48      = ws + 19456000;           //   reuse
  float* shaped32 = ws + 27136000;           // R3: 7,680,000
  float* shaped48 = ws + 27136000;
  float* cbuf     = ws + 34816000;           // R4: 5,120,000
  float* y32m     = ws + 34816000;           //   reuse
  float* kraw     = ws + 39936000;           // R5
  float* kfin     = ws + 39936000 + 1024000;
  _Float16* whh16 = (_Float16*)(ws + 39936000 + 1984000);

  // ---- feature net ----
  k_conv1<<<(32 * 128 * 500 + 255) / 256, 256, 0, stream>>>(features, c1w, c1b, c1out);
  k_conv2<<<(32 * 128 * 500 + 255) / 256, 256, 0, stream>>>(c1out, c2w, c2b, c2out);
  k_tconv<<<(32 * 128 * 1000 + 255) / 256, 256, 0, stream>>>(c2out, tw, tb, cbuf);

  // xw = cbuf @ wih^T + bih   (32000 x 384, K=128) via WMMA: 2000*24 tiles, 8 tile-waves/WG
  k_gemm_xw<<<(2000 * 24) / 8, 256, 0, stream>>>(cbuf, gwih, gbih, xw, 384, 24, 2000 * 24);

  // GRU (single workgroup, register-resident whh, WMMA recurrence)
  k_f32tof16<<<(384 * 128 + 255) / 256, 256, 0, stream>>>(gwhh, whh16, 384 * 128);
  k_gru<<<1, 384, 0, stream>>>(xw, whh16, gbhh, cf);

  // ---- 32 kHz path ----
  k_up2<<<(32 * 160000 + 255) / 256, 256, 0, stream>>>(x, up2_h, y32up);

  // af1 (1 -> 2)
  k_gemm_kgen<<<(2000 * 2) / 8, 256, 0, stream>>>(cf, af_wk[0], af_bk[0], af_wg[0], af_bg[0], 2, kraw, 2000 * 2);
  k_kpost<<<(32000 + 255) / 256, 256, 0, stream>>>(kraw, 2, 1, kfin);
  k_adaconv<<<32000, 256, 0, stream>>>(y32up, y32up, kfin, 1, 2, 160, 80, y32a0, y32a1);

  // td_shaper 1 on channel 1
  k_tdshaper<<<32000, 256, 0, stream>>>(y32a1, cf, ts1_w1, ts1_b1, ts1_w2, ts1_b2, 160, 8, shaped32);

  // af2 (2 -> 1)
  k_gemm_kgen<<<(2000 * 2) / 8, 256, 0, stream>>>(cf, af_wk[1], af_bk[1], af_wg[1], af_bg[1], 1, kraw, 2000 * 2);
  k_kpost<<<(32000 + 255) / 256, 256, 0, stream>>>(kraw, 1, 2, kfin);
  k_adaconv<<<32000, 256, 0, stream>>>(y32a0, shaped32, kfin, 2, 1, 160, 80, y32m, y32m);

  // ---- 48 kHz path ----
  k_up3half<<<(32 * 240000 + 255) / 256, 256, 0, stream>>>(y32m, up32_h, y48);

  // af3 (1 -> 2)
  k_gemm_kgen<<<(2000 * 2) / 8, 256, 0, stream>>>(cf, af_wk[2], af_bk[2], af_wg[2], af_bg[2], 2, kraw, 2000 * 2);
  k_kpost<<<(32000 + 255) / 256, 256, 0, stream>>>(kraw, 2, 1, kfin);
  k_adaconv<<<32000, 256, 0, stream>>>(y48, y48, kfin, 1, 2, 240, 120, y48a0, y48a1);

  // td_shaper 2 on channel 1
  k_tdshaper<<<32000, 256, 0, stream>>>(y48a1, cf, ts2_w1, ts2_b1, ts2_w2, ts2_b2, 240, 12, shaped48);

  // af4 (2 -> 1) -> final output
  k_gemm_kgen<<<(2000 * 2) / 8, 256, 0, stream>>>(cf, af_wk[3], af_bk[3], af_wg[3], af_bg[3], 1, kraw, 2000 * 2);
  k_kpost<<<(32000 + 255) / 256, 256, 0, stream>>>(kraw, 1, 2, kfin);
  k_adaconv<<<32000, 256, 0, stream>>>(y48a0, shaped48, kfin, 2, 1, 240, 120, dout, dout);
}
